// GATLayer_27195732918651
// MI455X (gfx1250) — matrix operand order repaired and checked by reading, not compiled, not run on hardware
//
#include <hip/hip_runtime.h>
#include <hip/hip_bf16.h>

// ---------------------------------------------------------------------------
// GAT layer for MI455X (gfx1250, wave32).
//   N=50000 nodes, E=800000 edges, IN_CH=256, H=4 heads, C=64 ch/head.
// Pipeline: bf16-split WMMA GEMM -> per-node attn scores -> 3 edge passes
// (segment max / exp-sum / weighted scatter) -> in-place ELU.
// ---------------------------------------------------------------------------

#define GN     50000
#define GE     800000
#define GIN    256
#define GH     4
#define GC     64
#define GHC    256   // H*C

typedef __attribute__((ext_vector_type(16))) __bf16 v16bf;
typedef __attribute__((ext_vector_type(8)))  float  v8f;

// Monotone float -> uint key for atomicMax-based float max.
__device__ __forceinline__ unsigned fkey(float f) {
    unsigned u = __float_as_uint(f);
    return (u & 0x80000000u) ? ~u : (u | 0x80000000u);
}
__device__ __forceinline__ float funkey(unsigned k) {
    return (k & 0x80000000u) ? __uint_as_float(k & 0x7fffffffu)
                             : __uint_as_float(~k);
}

// ---------------------------------------------------------------------------
// K0: W (fp32, [256][256] row-major, [m][k]) -> bf16 hi/lo split, same layout.
// This layout is used directly as the B operand per 16x16 tile, in contiguous k order like A:
// lane holds col m = m0+(lane&15), 16 contiguous bf16 at k0 + 16*(lane>>4).
// ---------------------------------------------------------------------------
__global__ void k_convert_w(const float* __restrict__ W,
                            __bf16* __restrict__ whi,
                            __bf16* __restrict__ wlo) {
    int t = blockIdx.x * blockDim.x + threadIdx.x;
    if (t >= GHC * GIN) return;
    float w = W[t];
    __bf16 hi = (__bf16)w;
    whi[t] = hi;
    wlo[t] = (__bf16)(w - (float)hi);
}

// ---------------------------------------------------------------------------
// K1: Wh = x @ W^T via v_wmma_f32_16x16x32_bf16, hi/lo precision split.
// One wave -> 16 rows x 64 cols (4 accumulators). 8 waves / block.
// ---------------------------------------------------------------------------
__global__ void __launch_bounds__(256)
k_gemm_wmma(const float* __restrict__ x,
            const __bf16* __restrict__ whi,
            const __bf16* __restrict__ wlo,
            float* __restrict__ Wh) {
    const int lane = threadIdx.x & 31;
    const int wave = threadIdx.x >> 5;
    const int gw   = blockIdx.x * 8 + wave;
    const int NWAVES = (GN / 16) * 4;          // 3125 row tiles * 4 col groups
    if (gw >= NWAVES) return;                   // wave-uniform: EXEC stays all-1s

    const int rt = gw >> 2;                     // row tile
    const int cg = gw & 3;                      // 64-col group
    const int n0 = rt * 16;
    const int m0 = cg * 64;
    const int hsel = lane >> 4;                 // 0: lanes 0-15, 1: lanes 16-31
    const int l15  = lane & 15;

    v8f acc[4] = {v8f{}, v8f{}, v8f{}, v8f{}};

    const float* xrow = x + (size_t)(n0 + l15) * GIN;

    for (int k0 = 0; k0 < GIN; k0 += 32) {
        __builtin_prefetch(xrow + k0 + 64, 0, 0);  // next K slab -> global_prefetch_b8

        // ---- A operand (16x32 bf16 layout): lane<16 -> K {0..15},
        //      lane>=16 -> K {16..31}, 2 bf16 per VGPR.
        const int ka = k0 + (hsel ? 16 : 0);
        float4 fa0 = *(const float4*)(xrow + ka);
        float4 fa1 = *(const float4*)(xrow + ka + 4);
        float4 fb0 = *(const float4*)(xrow + ka + 8);
        float4 fb1 = *(const float4*)(xrow + ka + 12);
        float f[16] = {fa0.x, fa0.y, fa0.z, fa0.w, fa1.x, fa1.y, fa1.z, fa1.w,
                       fb0.x, fb0.y, fb0.z, fb0.w, fb1.x, fb1.y, fb1.z, fb1.w};
        v16bf ahi, alo;
#pragma unroll
        for (int j = 0; j < 16; ++j) {
            __bf16 h = (__bf16)f[j];
            ahi[j] = h;
            alo[j] = (__bf16)(f[j] - (float)h);
        }

        // ---- B operands + 3-term WMMA per 16-col tile
        const int kb = k0 + (hsel ? 16 : 0);
#pragma unroll
        for (int ct = 0; ct < 4; ++ct) {
            const int m = m0 + ct * 16 + l15;
            v16bf bhi = *(const v16bf*)(whi + (size_t)m * GIN + kb);
            v16bf blo = *(const v16bf*)(wlo + (size_t)m * GIN + kb);
            acc[ct] = __builtin_amdgcn_wmma_f32_16x16x32_bf16(
                false, ahi, false, bhi, (short)0, acc[ct], false, false);
            acc[ct] = __builtin_amdgcn_wmma_f32_16x16x32_bf16(
                false, ahi, false, blo, (short)0, acc[ct], false, false);
            acc[ct] = __builtin_amdgcn_wmma_f32_16x16x32_bf16(
                false, alo, false, bhi, (short)0, acc[ct], false, false);
        }
    }

    // ---- Store: C/D layout — VGPR r: lanes 0-15 row r, lanes 16-31 row r+8.
#pragma unroll
    for (int ct = 0; ct < 4; ++ct) {
        const int col = m0 + ct * 16 + l15;
#pragma unroll
        for (int r = 0; r < 8; ++r) {
            const int row = n0 + (hsel ? r + 8 : r);
            Wh[(size_t)row * GHC + col] = acc[ct][r];
        }
    }
}

// ---------------------------------------------------------------------------
// K2: per-node attention scalars  s_src[n,h] = <Wh[n,h,:], att[h,:C]>,
//     s_dst[n,h] = <Wh[n,h,:], att[h,C:]>.
// ---------------------------------------------------------------------------
__global__ void k_scores(const float* __restrict__ Wh,
                         const float* __restrict__ att,   // [H][2C]
                         float* __restrict__ ssrc,
                         float* __restrict__ sdst) {
    int t = blockIdx.x * blockDim.x + threadIdx.x;
    if (t >= GN * GH) return;
    int n = t >> 2, h = t & 3;
    const float* w = Wh + (size_t)n * GHC + h * GC;
    const float* as = att + h * (2 * GC);
    const float* ad = as + GC;
    float s0 = 0.f, s1 = 0.f;
#pragma unroll 4
    for (int c = 0; c < GC; c += 4) {
        float4 v = *(const float4*)(w + c);
        float4 a = *(const float4*)(as + c);
        float4 b = *(const float4*)(ad + c);
        s0 += v.x * a.x + v.y * a.y + v.z * a.z + v.w * a.w;
        s1 += v.x * b.x + v.y * b.y + v.z * b.z + v.w * b.w;
    }
    ssrc[t] = s0;
    sdst[t] = s1;
}

// ---------------------------------------------------------------------------
// K3: zero-init output accumulator, denominators, max-keys (every call:
// harness does not re-poison between graph replays).
// ---------------------------------------------------------------------------
__global__ void k_init(float* __restrict__ out,
                       float* __restrict__ denom,
                       unsigned* __restrict__ emaxk) {
    int t = blockIdx.x * blockDim.x + threadIdx.x;
    if (t < GN * GHC) out[t] = 0.f;
    if (t < GN * GH) { denom[t] = 0.f; emaxk[t] = 0u; }
}

// ---------------------------------------------------------------------------
// K4: segment max of leaky_relu(s_src[src]+s_dst[dst]) over incoming edges.
// ---------------------------------------------------------------------------
__global__ void k_edge_max(const long long* __restrict__ ei,
                           const float* __restrict__ ssrc,
                           const float* __restrict__ sdst,
                           unsigned* __restrict__ emaxk) {
    int e = blockIdx.x * blockDim.x + threadIdx.x;
    if (e >= GE) return;
    int s = (int)ei[e];
    int d = (int)ei[GE + e];
    float4 a = *(const float4*)(ssrc + (size_t)s * GH);
    float4 b = *(const float4*)(sdst + (size_t)d * GH);
    float v[4] = {a.x + b.x, a.y + b.y, a.z + b.z, a.w + b.w};
#pragma unroll
    for (int h = 0; h < GH; ++h) {
        float lv = v[h] > 0.f ? v[h] : 0.2f * v[h];
        atomicMax(&emaxk[(size_t)d * GH + h], fkey(lv));
    }
}

// ---------------------------------------------------------------------------
// K5: e_exp = exp(e - max[dst]); denom[dst] += e_exp; stash e_exp per edge.
// ---------------------------------------------------------------------------
__global__ void k_edge_exp(const long long* __restrict__ ei,
                           const float* __restrict__ ssrc,
                           const float* __restrict__ sdst,
                           const unsigned* __restrict__ emaxk,
                           float* __restrict__ denom,
                           float* __restrict__ eexp) {
    int e = blockIdx.x * blockDim.x + threadIdx.x;
    if (e >= GE) return;
    int s = (int)ei[e];
    int d = (int)ei[GE + e];
    float4 a = *(const float4*)(ssrc + (size_t)s * GH);
    float4 b = *(const float4*)(sdst + (size_t)d * GH);
    float v[4] = {a.x + b.x, a.y + b.y, a.z + b.z, a.w + b.w};
#pragma unroll
    for (int h = 0; h < GH; ++h) {
        float lv = v[h] > 0.f ? v[h] : 0.2f * v[h];
        float m  = funkey(emaxk[(size_t)d * GH + h]);
        float ex = __expf(lv - m);
        eexp[(size_t)e * GH + h] = ex;
        atomicAdd(&denom[(size_t)d * GH + h], ex);
    }
}

// ---------------------------------------------------------------------------
// K6: weighted scatter-aggregate. One 256-thread block per edge:
// coalesced read of Wh[src,:], 256 no-return global f32 atomic adds.
// (Wh and out both fit MI455X's 192 MB L2 -> scatter stays on-chip.)
// ---------------------------------------------------------------------------
__global__ void __launch_bounds__(256)
k_aggregate(const long long* __restrict__ ei,
            const float* __restrict__ Wh,
            const float* __restrict__ eexp,
            const float* __restrict__ denom,
            float* __restrict__ out) {
    int e = blockIdx.x;
    int c = threadIdx.x;        // 0..255
    int h = c >> 6;             // head
    int s = (int)ei[e];
    int d = (int)ei[GE + e];
    float alpha = eexp[(size_t)e * GH + h] /
                  (denom[(size_t)d * GH + h] + 1e-16f);
    atomicAdd(&out[(size_t)d * GHC + c], alpha * Wh[(size_t)s * GHC + c]);
}

// ---------------------------------------------------------------------------
// K7: in-place ELU finalize.
// ---------------------------------------------------------------------------
__global__ void k_elu(float* __restrict__ out) {
    int t = blockIdx.x * blockDim.x + threadIdx.x;
    if (t >= GN * GHC) return;
    float v = out[t];
    out[t] = v > 0.f ? v : expm1f(v);
}

// ---------------------------------------------------------------------------
extern "C" void kernel_launch(void* const* d_in, const int* in_sizes, int n_in,
                              void* d_out, int out_size, void* d_ws, size_t ws_size,
                              hipStream_t stream) {
    const float*     x   = (const float*)d_in[0];       // [N, 256]
    const long long* ei  = (const long long*)d_in[1];   // [2, E] int64
    const float*     W   = (const float*)d_in[2];       // [256, 256]
    const float*     att = (const float*)d_in[3];       // [4, 128]
    float* out = (float*)d_out;                          // [N, 256]

    // Workspace carve-up (~67 MB total).
    char* base = (char*)d_ws;
    size_t off = 0;
    auto take = [&](size_t nbytes) -> char* {
        char* p = base + off;
        off += (nbytes + 255) & ~(size_t)255;
        return p;
    };
    __bf16*   whi   = (__bf16*)take((size_t)GHC * GIN * sizeof(__bf16));
    __bf16*   wlo   = (__bf16*)take((size_t)GHC * GIN * sizeof(__bf16));
    float*    Wh    = (float*)take((size_t)GN * GHC * sizeof(float));
    float*    ssrc  = (float*)take((size_t)GN * GH * sizeof(float));
    float*    sdst  = (float*)take((size_t)GN * GH * sizeof(float));
    unsigned* emaxk = (unsigned*)take((size_t)GN * GH * sizeof(unsigned));
    float*    denom = (float*)take((size_t)GN * GH * sizeof(float));
    float*    eexp  = (float*)take((size_t)GE * GH * sizeof(float));
    (void)ws_size; (void)in_sizes; (void)n_in; (void)out_size;

    // K0: W -> bf16 hi/lo
    k_convert_w<<<(GHC * GIN + 255) / 256, 256, 0, stream>>>(W, whi, wlo);

    // K1: GEMM (3125 row tiles x 4 col groups = 12500 waves, 8 waves/block)
    {
        int nwaves = (GN / 16) * 4;
        int blocks = (nwaves + 7) / 8;
        k_gemm_wmma<<<blocks, 256, 0, stream>>>(x, whi, wlo, Wh);
    }

    // K2: per-node attention scalars
    k_scores<<<(GN * GH + 255) / 256, 256, 0, stream>>>(Wh, att, ssrc, sdst);

    // K3: init accumulators
    k_init<<<(GN * GHC + 255) / 256, 256, 0, stream>>>(out, denom, emaxk);

    // K4: segment max
    k_edge_max<<<(GE + 255) / 256, 256, 0, stream>>>(ei, ssrc, sdst, emaxk);

    // K5: exp + denominator
    k_edge_exp<<<(GE + 255) / 256, 256, 0, stream>>>(ei, ssrc, sdst, emaxk,
                                                     denom, eexp);

    // K6: weighted aggregation (one block per edge)
    k_aggregate<<<GE, 256, 0, stream>>>(ei, Wh, eexp, denom, out);

    // K7: ELU
    k_elu<<<(GN * GHC + 255) / 256, 256, 0, stream>>>(out);
}
